// AttentionModel_49873160241285
// MI455X (gfx1250) — matrix operand order repaired and checked by reading, not compiled, not run on hardware
//
#include <hip/hip_runtime.h>
#include <cstdint>

// Problem constants (from reference)
#define BATCH 64
#define LSEQ  4096
#define DIN   512
#define NOUT  256

typedef __attribute__((ext_vector_type(16))) __bf16 v16bf;
typedef __attribute__((ext_vector_type(8)))  __bf16 v8bf;
typedef __attribute__((ext_vector_type(8)))  float  v8f;

// ---------------------------------------------------------------------------
// ws layout (bytes):
//   [0,       262144)  Wp : packed bf16 B-fragments, 16 tiles x 16 ksteps x 32 lanes x 16 elems
//   [262144,  327680)  qlin : [64,256] f32
//   [327680, 1376256)  e    : [64,4096] f32
// ---------------------------------------------------------------------------

__device__ __forceinline__ uint32_t pk_bf16(float a, float b) {
    uint32_t ua = __float_as_uint(a), ub = __float_as_uint(b);
    ua = (ua + 0x7FFFu + ((ua >> 16) & 1u)) >> 16;   // round-to-nearest-even
    ub = (ub + 0x7FFFu + ((ub >> 16) & 1u)) >> 16;
    return ua | (ub << 16);
}

// Repack W_para [256,512] f32 -> bf16 in the wave32 WMMA B-matrix fragment order:
// entry index g = ((t*16 + ks)*32 + lane)*16 + e ; N=t*16+lane%16, h=lane/16:
//   e<8: K=ks*32+8h+e ; e>=8: K=ks*32+16+8h+(e-8)
__global__ void pack_w_kernel(const float* __restrict__ W, uint16_t* __restrict__ Wp) {
    int g    = blockIdx.x * 256 + threadIdx.x;   // 131072 total
    int ei   = g & 15;
    int lane = (g >> 4) & 31;
    int ks   = (g >> 9) & 15;
    int t    = g >> 13;
    int o    = t * 16 + (lane & 15);
    int k    = ks * 32 + 8 * (lane >> 4) + (ei & 7) + ((ei & 8) << 1);
    uint32_t u = __float_as_uint(W[o * DIN + k]);
    u = (u + 0x7FFFu + ((u >> 16) & 1u)) >> 16;
    Wp[g] = (uint16_t)u;
}

// q_lin[b,o] = query[b,:] . W_query[o,:] + b_query[o]
__global__ void qlin_kernel(const float* __restrict__ query, const float* __restrict__ Wq,
                            const float* __restrict__ bq, float* __restrict__ qlin) {
    int b = blockIdx.x;
    int o = threadIdx.x;                         // 256 threads
    __shared__ __align__(16) float qs[DIN];
    for (int i = threadIdx.x; i < DIN; i += 256) qs[i] = query[b * DIN + i];
    __syncthreads();
    const float4* w  = (const float4*)(Wq + (size_t)o * DIN);
    const float4* q4 = (const float4*)qs;
    float acc = bq[o];
#pragma unroll 8
    for (int i = 0; i < DIN / 4; ++i) {
        float4 wv = w[i], qv = q4[i];
        acc += wv.x * qv.x + wv.y * qv.y + wv.z * qv.z + wv.w * qv.w;
    }
    qlin[b * NOUT + o] = acc;
}

// Fused GEMM(bf16 WMMA) + tanh + dot(attn_vec) -> e[b,l]
// block = 256 threads (8 waves), tile = 32 rows x 256 cols, K = 512
__launch_bounds__(256)
__global__ void attn_score_kernel(const float* __restrict__ para,
                                  const uint16_t* __restrict__ Wp,
                                  const float* __restrict__ qlin,
                                  const float* __restrict__ avec,
                                  float* __restrict__ e_out) {
    __shared__ __align__(16) uint32_t Atile[32 * 256];  // 32 rows x 512 bf16 = 32 KB
    __shared__ float etile[32];

    const int tid  = threadIdx.x;
    const int row0 = blockIdx.x * 32;            // 4096 % 32 == 0 -> single batch per block
    const int b    = row0 >> 12;

    // ---- stage A tile: fp32 -> bf16 into LDS (each thread: one row, 64 floats) ----
    {
        const int r = tid >> 3;                  // 0..31
        const int c = tid & 7;                   // 0..7 (64 floats each)
        const float4* src = (const float4*)(para + (size_t)(row0 + r) * DIN + c * 64);
        uint32_t* dst = &Atile[r * 256 + c * 32];
#pragma unroll
        for (int i = 0; i < 16; ++i) {
            float4 v = src[i];
            dst[i * 2 + 0] = pk_bf16(v.x, v.y);
            dst[i * 2 + 1] = pk_bf16(v.z, v.w);
        }
    }
    if (tid < 32) etile[tid] = 0.0f;
    __syncthreads();

    const int lane = tid & 31;
    const int wave = tid >> 5;
    const int hl   = lane >> 4;                  // K-half selector per ISA A/B layouts
    const int m    = lane & 15;

    const __bf16* A = (const __bf16*)Atile;
    const int t0 = wave * 2, t1 = t0 + 1;        // two N-tiles per wave

    v8f acc00 = {}; v8f acc01 = {}; v8f acc10 = {}; v8f acc11 = {};

    for (int ks = 0; ks < 16; ++ks) {
        const int k0 = ks * 32 + (hl ? 8 : 0);
        // A fragments per the 16-bit 16x32 layout: elems 0..7 -> K=k0.., 8..15 -> K=k0+16..
        const __bf16* a0p = A + (m * 512 + k0);          // rows 0..15
        const __bf16* a1p = A + ((16 + m) * 512 + k0);   // rows 16..31
        v8bf a0lo = *(const v8bf*)a0p;
        v8bf a0hi = *(const v8bf*)(a0p + 16);
        v8bf a1lo = *(const v8bf*)a1p;
        v8bf a1hi = *(const v8bf*)(a1p + 16);
        v16bf a0 = __builtin_shufflevector(a0lo, a0hi, 0,1,2,3,4,5,6,7,8,9,10,11,12,13,14,15);
        v16bf a1 = __builtin_shufflevector(a1lo, a1hi, 0,1,2,3,4,5,6,7,8,9,10,11,12,13,14,15);
        // B fragments: contiguous 32 B per lane from the prepacked buffer
        const v8bf* w0 = (const v8bf*)(Wp + (((size_t)(t0 * 16 + ks) * 32 + lane) << 4));
        const v8bf* w1 = (const v8bf*)(Wp + (((size_t)(t1 * 16 + ks) * 32 + lane) << 4));
        v16bf b0 = __builtin_shufflevector(w0[0], w0[1], 0,1,2,3,4,5,6,7,8,9,10,11,12,13,14,15);
        v16bf b1 = __builtin_shufflevector(w1[0], w1[1], 0,1,2,3,4,5,6,7,8,9,10,11,12,13,14,15);

        acc00 = __builtin_amdgcn_wmma_f32_16x16x32_bf16(false, a0, false, b0, (short)0, acc00, false, false);
        acc01 = __builtin_amdgcn_wmma_f32_16x16x32_bf16(false, a0, false, b1, (short)0, acc01, false, false);
        acc10 = __builtin_amdgcn_wmma_f32_16x16x32_bf16(false, a1, false, b0, (short)0, acc10, false, false);
        acc11 = __builtin_amdgcn_wmma_f32_16x16x32_bf16(false, a1, false, b1, (short)0, acc11, false, false);
    }

    // ---- epilogue: tanh(d + qlin[o]) * avec[o], reduce 16 columns per tile ----
    // D layout: VGPR j -> M=j (lanes 0..15) / M=8+j (lanes 16..31), N = lane%16
    auto reduce_acc = [&](v8f acc, int tl, int mbase) {
        const int o  = tl * 16 + m;
        const float ql = qlin[b * NOUT + o];
        const float av = avec[o];
#pragma unroll
        for (int j = 0; j < 8; ++j) {
            float v = tanhf(acc[j] + ql) * av;
            v += __shfl_xor(v, 1, 32);
            v += __shfl_xor(v, 2, 32);
            v += __shfl_xor(v, 4, 32);
            v += __shfl_xor(v, 8, 32);           // full sum within each 16-lane column group
            if (m == 0) atomicAdd(&etile[mbase + hl * 8 + j], v);
        }
    };
    reduce_acc(acc00, t0, 0);
    reduce_acc(acc01, t1, 0);
    reduce_acc(acc10, t0, 16);
    reduce_acc(acc11, t1, 16);
    __syncthreads();
    if (tid < 32) e_out[(size_t)row0 + tid] = etile[tid];
}

// Masked softmax + renorm. Algebra: out = exp(e-M)*mask / sum(exp(e-M)*mask), or 0.
__global__ void softmax_kernel(const float* __restrict__ e, const int* __restrict__ mask,
                               float* __restrict__ out) {
    const int b = blockIdx.x, tid = threadIdx.x; // 256 threads, 16 elems each
    __shared__ float red[256];
    float ev[16], mk[16];
    float mx = -3.402823466e38f;
#pragma unroll
    for (int i = 0; i < 16; ++i) {
        int idx = i * 256 + tid;
        ev[i] = e[(size_t)b * LSEQ + idx];
        mk[i] = (float)mask[(size_t)b * LSEQ + idx];
        mx = fmaxf(mx, ev[i]);
    }
    red[tid] = mx; __syncthreads();
    for (int s = 128; s > 0; s >>= 1) { if (tid < s) red[tid] = fmaxf(red[tid], red[tid + s]); __syncthreads(); }
    mx = red[0]; __syncthreads();

    float pv[16], sum = 0.0f;
#pragma unroll
    for (int i = 0; i < 16; ++i) { pv[i] = __expf(ev[i] - mx) * mk[i]; sum += pv[i]; }
    red[tid] = sum; __syncthreads();
    for (int s = 128; s > 0; s >>= 1) { if (tid < s) red[tid] += red[tid + s]; __syncthreads(); }
    const float S = red[0];
    const float inv = (S != 0.0f) ? (1.0f / S) : 0.0f;
#pragma unroll
    for (int i = 0; i < 16; ++i) out[(size_t)b * LSEQ + i * 256 + tid] = pv[i] * inv;
}

extern "C" void kernel_launch(void* const* d_in, const int* in_sizes, int n_in,
                              void* d_out, int out_size, void* d_ws, size_t ws_size,
                              hipStream_t stream) {
    const float* para   = (const float*)d_in[0];  // [64,4096,512] f32
    const float* query  = (const float*)d_in[1];  // [64,512] f32
    const int*   maskp  = (const int*)  d_in[2];  // [64,4096] i32
    const float* Wpara  = (const float*)d_in[3];  // [256,512] f32
    const float* Wquery = (const float*)d_in[4];  // [256,512] f32
    const float* bq     = (const float*)d_in[5];  // [256] f32
    const float* avec   = (const float*)d_in[6];  // [256] f32
    float* out = (float*)d_out;                   // [64,4096] f32

    uint16_t* Wp   = (uint16_t*)d_ws;                              // 256 KB
    float*    qlin = (float*)((char*)d_ws + 262144);               // 64 KB
    float*    ebuf = (float*)((char*)d_ws + 262144 + 65536);       // 1 MB

    pack_w_kernel<<<131072 / 256, 256, 0, stream>>>(Wpara, Wp);
    qlin_kernel<<<BATCH, 256, 0, stream>>>(query, Wquery, bq, qlin);
    attn_score_kernel<<<(BATCH * LSEQ) / 32, 256, 0, stream>>>(para, Wp, qlin, avec, ebuf);
    softmax_kernel<<<BATCH, 256, 0, stream>>>(ebuf, maskp, out);
}
